// LST_23940147708154
// MI455X (gfx1250) — matrix operand order repaired and checked by reading, not compiled, not run on hardware
//
#include <hip/hip_runtime.h>
#include <hip/hip_bf16.h>
#include <math.h>

// ---------------------------------------------------------------------------
// GST cross-attention for MI455X (gfx1250, wave32, WMMA f16->f32 + TDM).
//   prep kernel (1 block): build pre-swizzled f16 WMMA fragments in ws:
//       Wq A-fragments, kh = tanh(embed)@Wk^T * 1/sqrt(32), vh = tanh@Wv^T
//   main kernel: TDM tensor_load_to_lds stages the 152KB fragment image,
//       then per 16-row tile: q^T = Wq @ x^T (16x16x32 f16 WMMA, K=256),
//       logits^T = kh @ q^T, softmax over 10 tokens, out^T = vh^T @ attn^T.
// ---------------------------------------------------------------------------

typedef _Float16 v16h __attribute__((ext_vector_type(16)));
typedef float    v8f  __attribute__((ext_vector_type(8)));
typedef unsigned int v4u __attribute__((ext_vector_type(4)));
typedef int          v4i __attribute__((ext_vector_type(4)));
typedef int          v8i __attribute__((ext_vector_type(8)));

#define N_TOK     10
#define NHEAD     8
#define DHEAD     32
#define DQ        256
#define UNITS     256
#define T_ALL     2000
#define N_BATCH   64
#define M_TOTAL   (N_BATCH * T_ALL)     // 128000 query rows
#define ROWS_PER_BLOCK 128              // 8 waves * 16 rows

// Fragment image layout (in halfs): Wq frags, kh frags, vh frags
#define WQ_HALFS  (16 * 8 * 32 * 16)    // [utile][kfrag][lane][16] = 65536
#define KH_HALFS  (8 * 32 * 16)         // [head][lane][16]         = 4096
#define VH_HALFS  (8 * 2 * 32 * 16)     // [head][dtile][lane][16]  = 8192
#define FRAG_HALFS (WQ_HALFS + KH_HALFS + VH_HALFS)      // 77824
#define SMEM_BYTES (FRAG_HALFS * 2)                      // 155648 B
#define FRAG_QW    (SMEM_BYTES / 8)                      // 19456 8-byte elems

static __device__ __forceinline__ v8f wmma16(v16h a, v16h b, v8f c) {
  // D = A(16x32 f16) * B(32x16 f16) + C(16x16 f32)
  return __builtin_amdgcn_wmma_f32_16x16x32_f16(false, a, false, b,
                                                (short)0, c, false, false);
}

// ---------------------------------------------------------------------------
// Prologue (1 block, 256 threads): build the fragment image in workspace.
// ---------------------------------------------------------------------------
__global__ void __launch_bounds__(256, 1)
gst_prep_kernel(const float* __restrict__ embed,   // [10][32]
                const float* __restrict__ Wq,      // [256][256]
                const float* __restrict__ Wk,      // [256][32]
                const float* __restrict__ Wv,      // [256][32]
                _Float16* __restrict__ wsF)
{
  __shared__ float tanhE[N_TOK * DHEAD];
  const int tid = threadIdx.x;
  for (int i = tid; i < N_TOK * DHEAD; i += 256) tanhE[i] = tanhf(embed[i]);
  __syncthreads();

  _Float16* wqF = wsF;
  _Float16* khF = wqF + WQ_HALFS;
  _Float16* vhF = khF + KH_HALFS;

  // Wq A-fragments: fid = (utile<<8)|(kfrag<<5)|lane, 16 f16 each
  for (int fid = tid; fid < 16 * 8 * 32; fid += 256) {
    const int utile = fid >> 8, kf = (fid >> 5) & 7, ln = fid & 31;
    const int u  = utile * 16 + (ln & 15);
    const int kb = kf * 32 + ((ln >> 4) << 4);
    const float* wr = Wq + (size_t)u * DQ + kb;
    const float4 w0 = *(const float4*)(wr);
    const float4 w1 = *(const float4*)(wr + 4);
    const float4 w2 = *(const float4*)(wr + 8);
    const float4 w3 = *(const float4*)(wr + 12);
    v16h fv;
    fv[0]=(_Float16)w0.x; fv[1]=(_Float16)w0.y; fv[2]=(_Float16)w0.z; fv[3]=(_Float16)w0.w;
    fv[4]=(_Float16)w1.x; fv[5]=(_Float16)w1.y; fv[6]=(_Float16)w1.z; fv[7]=(_Float16)w1.w;
    fv[8]=(_Float16)w2.x; fv[9]=(_Float16)w2.y; fv[10]=(_Float16)w2.z; fv[11]=(_Float16)w2.w;
    fv[12]=(_Float16)w3.x; fv[13]=(_Float16)w3.y; fv[14]=(_Float16)w3.z; fv[15]=(_Float16)w3.w;
    *(v16h*)(wqF + (size_t)fid * 16) = fv;
  }

  // kh A-fragments: M=tok(pad 16), K=32(d); 1/sqrt(32) folded in
  {
    const int h = tid >> 5, ln = tid & 31;
    const int tok = ln & 15, grp = ln >> 4;
    v16h kv = {};
    if (tok < N_TOK) {
      for (int i = 0; i < 16; ++i) {
        const int d = grp * 16 + i;
        const int u = h * DHEAD + d;
        float s = 0.f;
        for (int j = 0; j < DHEAD; ++j)
          s += tanhE[tok * DHEAD + j] * Wk[(size_t)u * DHEAD + j];
        kv[i] = (_Float16)(s * 0.17677669529663689f);   // 1/sqrt(32)
      }
    }
    *(v16h*)(khF + (size_t)tid * 16) = kv;
  }

  // vh A-fragments: M=d(16 per dtile), K=tok(pad 32 with zeros)
  for (int r = 0; r < 2; ++r) {
    const int fid = tid + r * 256;                  // 0..511
    const int h = fid >> 6, dt = (fid >> 5) & 1, ln = fid & 31;
    const int dl = ln & 15, grp = ln >> 4;
    v16h vv = {};
    for (int i = 0; i < 16; ++i) {
      const int tok = grp * 16 + i;
      if (tok < N_TOK) {
        const int u = h * DHEAD + dt * 16 + dl;
        float s = 0.f;
        for (int j = 0; j < DHEAD; ++j)
          s += tanhE[tok * DHEAD + j] * Wv[(size_t)u * DHEAD + j];
        vv[i] = (_Float16)s;
      }
    }
    *(v16h*)(vhF + (size_t)fid * 16) = vv;
  }
}

// ---------------------------------------------------------------------------
// Main kernel: 1000 blocks x 256 threads; 152KB LDS staged via TDM.
// ---------------------------------------------------------------------------
__global__ void __launch_bounds__(256, 1)
gst_attn_kernel(const float* __restrict__ inputs,
                const int*   __restrict__ pidx,        // positional_indexes
                const float* __restrict__ ptab,        // [2001][256]
                const _Float16* __restrict__ wsF,      // fragment image
                float* __restrict__ out,               // [N*T][256]
                float* __restrict__ scores)            // [N][80][T]
{
  extern __shared__ char smem[];
  const _Float16* wqF = (const _Float16*)smem;
  const _Float16* khF = wqF + WQ_HALFS;
  const _Float16* vhF = khF + KH_HALFS;

  const int tid  = threadIdx.x;
  const int wid  = tid >> 5;
  const int lane = tid & 31;

  // ---- stage fragment image ws -> LDS via the Tensor Data Mover ---------
#if __has_builtin(__builtin_amdgcn_tensor_load_to_lds)
  if (wid == 0) {
    // Tensor DMA descriptor (D#): 1-D copy, 8-byte elements, FRAG_QW elems.
    const unsigned long long ga = (unsigned long long)wsF;
    v4u g0;
    g0[0] = 1u;                                        // count=1, user mode
    g0[1] = 0u;                                        // lds_addr = 0
    g0[2] = (unsigned)(ga & 0xFFFFFFFFull);            // global_addr[31:0]
    g0[3] = (unsigned)((ga >> 32) & 0x1FFFFFFull)      // global_addr[56:32]
            | 0x80000000u;                             // type=2 ("image")
    v8i g1;
    g1[0] = 0x00030000;                                // data_size=3 (8B)
    g1[1] = (int)((FRAG_QW & 0xFFFF) << 16);           // tensor_dim0[15:0]
    g1[2] = (int)(((FRAG_QW >> 16) & 0xFFFF) | (1u << 16)); // dim0 hi | tensor_dim1=1
    g1[3] = (int)((FRAG_QW & 0xFFFF) << 16);           // tile_dim0
    g1[4] = 0;                                         // tile_dim1=0, tile_dim2=0
    g1[5] = (int)FRAG_QW;                              // tensor_dim0_stride lo
    g1[6] = 0;
    g1[7] = 0;
    v4i g2 = {0, 0, 0, 0};
    v4i g3 = {0, 0, 0, 0};
    v8i g4 = {0, 0, 0, 0, 0, 0, 0, 0};                 // extra group (clang-23)
    __builtin_amdgcn_tensor_load_to_lds(g0, g1, g2, g3, g4, 0);
    __builtin_amdgcn_s_wait_tensorcnt(0);
  }
#else
  {
    const float4* src = (const float4*)wsF;
    float4* dst = (float4*)smem;
    for (int i = tid; i < SMEM_BYTES / 16; i += 256) dst[i] = src[i];
  }
#endif
  __syncthreads();

  // ---- each wave owns 16 query rows (t values) --------------------------
  const int tl   = lane & 15;        // local t
  const int grp  = lane >> 4;        // 0 or 1
  const bool top = (grp != 0);

  const long m    = (long)blockIdx.x * ROWS_PER_BLOCK + wid * 16 + tl;
  const int n_idx = (int)(m / T_ALL);
  const int t     = (int)(m % T_ALL);

  const int   pix = pidx[m];
  const float* xr = inputs + m * (long)DQ;
  const float* pr = ptab + (long)pix * DQ;

  // B fragments of x^T: lane holds 16 consecutive K at fixed t (gather+add fused)
  v16h bx[8];
#pragma unroll
  for (int f = 0; f < 8; ++f) {
    const int kb = f * 32 + grp * 16;
    const float4 a0 = *(const float4*)(xr + kb);
    const float4 a1 = *(const float4*)(xr + kb + 4);
    const float4 a2 = *(const float4*)(xr + kb + 8);
    const float4 a3 = *(const float4*)(xr + kb + 12);
    const float4 p0 = *(const float4*)(pr + kb);
    const float4 p1 = *(const float4*)(pr + kb + 4);
    const float4 p2 = *(const float4*)(pr + kb + 8);
    const float4 p3 = *(const float4*)(pr + kb + 12);
    v16h b;
    b[0]=(_Float16)(a0.x+p0.x); b[1]=(_Float16)(a0.y+p0.y); b[2]=(_Float16)(a0.z+p0.z); b[3]=(_Float16)(a0.w+p0.w);
    b[4]=(_Float16)(a1.x+p1.x); b[5]=(_Float16)(a1.y+p1.y); b[6]=(_Float16)(a1.z+p1.z); b[7]=(_Float16)(a1.w+p1.w);
    b[8]=(_Float16)(a2.x+p2.x); b[9]=(_Float16)(a2.y+p2.y); b[10]=(_Float16)(a2.z+p2.z); b[11]=(_Float16)(a2.w+p2.w);
    b[12]=(_Float16)(a3.x+p3.x); b[13]=(_Float16)(a3.y+p3.y); b[14]=(_Float16)(a3.z+p3.z); b[15]=(_Float16)(a3.w+p3.w);
    bx[f] = b;
  }

  float* orow = out + m * (long)UNITS;
  const v8f zero8 = {};

  for (int h = 0; h < NHEAD; ++h) {
    // q^T tiles for this head: u-tiles 2h and 2h+1 (d = 0..15 and 16..31)
    v8f c0 = zero8, c1 = zero8;
#pragma unroll
    for (int f = 0; f < 8; ++f) {
      const v16h a = *(const v16h*)(wqF + ((size_t)((2*h) * 8 + f) * 32 + lane) * 16);
      c0 = wmma16(a, bx[f], c0);
    }
#pragma unroll
    for (int f = 0; f < 8; ++f) {
      const v16h a = *(const v16h*)(wqF + ((size_t)((2*h+1) * 8 + f) * 32 + lane) * 16);
      c1 = wmma16(a, bx[f], c1);
    }

    // Build B fragment of q_head^T: lane needs 16 consecutive d at fixed t.
    // Rows 8..15 of each u-tile live in lane^16 -> one xor-16 swap per reg.
    v16h qf;
#pragma unroll
    for (int j = 0; j < 8; ++j) {
      const float s0 = __shfl_xor(c0[j], 16);
      const float s1 = __shfl_xor(c1[j], 16);
      qf[j]     = (_Float16)(top ? s1 : c0[j]);
      qf[j + 8] = (_Float16)(top ? c1[j] : s0);
    }

    // logits^T[tok][t] (scale pre-folded into kh)
    const v16h ka = *(const v16h*)(khF + ((size_t)h * 32 + lane) * 16);
    const v8f lg = wmma16(ka, qf, zero8);

    // softmax over 10 tokens: lane group0 holds tok 0..7, group1 tok 8..9
    float pmax = -3.0e38f;
#pragma unroll
    for (int j = 0; j < 8; ++j) {
      const float v = (top && j >= 2) ? -3.0e38f : lg[j];
      pmax = fmaxf(pmax, v);
    }
    const float omax = fmaxf(pmax, __shfl_xor(pmax, 16));
    float aP[8]; float psum = 0.f;
#pragma unroll
    for (int j = 0; j < 8; ++j) {
      const float v = (top && j >= 2) ? 0.f : __expf(lg[j] - omax);
      aP[j] = v; psum += v;
    }
    const float osum = psum + __shfl_xor(psum, 16);
    const float inv  = 1.0f / osum;
#pragma unroll
    for (int j = 0; j < 8; ++j) aP[j] *= inv;

    // scores: [n, h*10+tok, t]
#pragma unroll
    for (int j = 0; j < 8; ++j) {
      const int tok = grp * 8 + j;
      if (tok < N_TOK)
        scores[((long)n_idx * (NHEAD * N_TOK) + h * N_TOK + tok) * T_ALL + t] = aP[j];
    }

    // B fragment of attn^T: lane group0 -> tok 0..15 (8 own + 8 swapped),
    // group1 covers tok 16..31 which are all zero padding.
    v16h af;
#pragma unroll
    for (int j = 0; j < 8; ++j) {
      const float sa = __shfl_xor(aP[j], 16);
      af[j]     = top ? (_Float16)0.0f : (_Float16)aP[j];
      af[j + 8] = top ? (_Float16)0.0f : (_Float16)sa;
    }

    // out^T[d][t] = vh^T @ attn^T, two d-tiles of 16
#pragma unroll
    for (int dt = 0; dt < 2; ++dt) {
      const v16h va = *(const v16h*)(vhF + ((size_t)(h * 2 + dt) * 32 + lane) * 16);
      const v8f o = wmma16(va, af, zero8);
      float* p = orow + h * DHEAD + dt * 16 + grp * 8;  // 8 consecutive d
      float4 lo, hi4;
      lo.x = o[0]; lo.y = o[1]; lo.z = o[2]; lo.w = o[3];
      hi4.x = o[4]; hi4.y = o[5]; hi4.z = o[6]; hi4.w = o[7];
      *(float4*)p = lo;
      *(float4*)(p + 4) = hi4;
    }
  }
}

extern "C" void kernel_launch(void* const* d_in, const int* in_sizes, int n_in,
                              void* d_out, int out_size, void* d_ws, size_t ws_size,
                              hipStream_t stream) {
  (void)in_sizes; (void)n_in; (void)out_size; (void)ws_size;
  const float* inputs = (const float*)d_in[0];
  const int*   pidx   = (const int*)d_in[1];    // harness: integer -> const int*
  const float* embed  = (const float*)d_in[2];
  const float* Wq     = (const float*)d_in[3];
  const float* Wk     = (const float*)d_in[4];
  const float* Wv     = (const float*)d_in[5];
  const float* ptab   = (const float*)d_in[6];
  float* out    = (float*)d_out;
  float* scores = out + (size_t)M_TOTAL * UNITS;  // outputs concatenated flat
  _Float16* wsF = (_Float16*)d_ws;

  // 1) build fragment image once
  gst_prep_kernel<<<dim3(1), dim3(256), 0, stream>>>(embed, Wq, Wk, Wv, wsF);

  // 2) main attention kernel
  (void)hipFuncSetAttribute((const void*)gst_attn_kernel,
                            hipFuncAttributeMaxDynamicSharedMemorySize, SMEM_BYTES);
  dim3 grid(M_TOTAL / ROWS_PER_BLOCK), block(256);
  gst_attn_kernel<<<grid, block, SMEM_BYTES, stream>>>(
      inputs, pidx, ptab, wsF, out, scores);
}
